// VariationalEncoder_53661321396311
// MI455X (gfx1250) — compile-verified, matrix operand order checked by reading, not run on hardware
//
#include <hip/hip_runtime.h>

typedef __attribute__((ext_vector_type(16))) __bf16 v16bf;
typedef __attribute__((ext_vector_type(8)))  float  v8f;

#define NNODES 50000
#define NEDGES 800000

#define MODE_DUAL_LN 0
#define MODE_ADD_LN  1
#define MODE_DUAL    2

// ---------------------------------------------------------------------------
// fp32 -> bf16 helpers (round to nearest even)
// ---------------------------------------------------------------------------
__device__ __forceinline__ unsigned bf_bits(float f) {
  unsigned u = __builtin_bit_cast(unsigned, f);
  u += 0x7FFFu + ((u >> 16) & 1u);
  return u >> 16;
}
__device__ __forceinline__ __bf16 f2bf(float f) {
  unsigned short h = (unsigned short)bf_bits(f);
  return __builtin_bit_cast(__bf16, h);
}
__device__ __forceinline__ unsigned long long pack4bf(float4 v) {
  return (unsigned long long)bf_bits(v.x) |
         ((unsigned long long)bf_bits(v.y) << 16) |
         ((unsigned long long)bf_bits(v.z) << 32) |
         ((unsigned long long)bf_bits(v.w) << 48);
}

// ---------------------------------------------------------------------------
// zero a buffer (float4 granularity; all sizes here are multiples of 4)
// ---------------------------------------------------------------------------
__global__ void zero_f4(float4* __restrict__ p, long long n4) {
  long long i = (long long)blockIdx.x * blockDim.x + threadIdx.x;
  if (i < n4) {
    float4 z; z.x = 0.f; z.y = 0.f; z.z = 0.f; z.w = 0.f;
    p[i] = z;
  }
}

// ---------------------------------------------------------------------------
// edge aggregation: agg[dst] += x[src]   (segment_sum, fp32 atomics at L2)
// blockDim = (F/4, 256/(F/4)); no integer division in the hot path.
// ---------------------------------------------------------------------------
template <int F>
__global__ void scatter_add4(const float* __restrict__ x,
                             const long long* __restrict__ src,
                             const long long* __restrict__ dst,
                             float* __restrict__ agg) {
  constexpr int G   = F / 4;     // float4 groups per edge (32 or 64)
  constexpr int EPB = 256 / G;   // edges per block (8 or 4)
  const int g = threadIdx.x << 2;
  long long e = (long long)blockIdx.x * EPB + threadIdx.y;
  long long s = src[e], d = dst[e];
  const float4 v = *reinterpret_cast<const float4*>(x + s * F + g);
  float* o = agg + d * F + g;
  atomicAdd(o + 0, v.x);
  atomicAdd(o + 1, v.y);
  atomicAdd(o + 2, v.z);
  atomicAdd(o + 3, v.w);
}

// ---------------------------------------------------------------------------
// pre-pack fp32 weight [Fin, Fout] into bf16 WMMA B-fragment order:
//   dst[((ct*ksteps + kt)*32 + lane)*16 + e]
// lane 0-15  = col, elems 0..7 -> K 0..7,  elems 8..15 -> K 16..23
// lane 16-31 = col, elems 0..7 -> K 8..15, elems 8..15 -> K 24..31
// ---------------------------------------------------------------------------
__global__ void prepack_w(const float* __restrict__ W, __bf16* __restrict__ dst,
                          int Fin, int Fout) {
  int idx = blockIdx.x * blockDim.x + threadIdx.x;
  if (idx >= Fin * Fout) return;
  int e    = idx & 15;
  int frag = idx >> 4;
  int lane = frag & 31;
  int t    = frag >> 5;
  int ksteps = Fin >> 5;
  int kt = t % ksteps;
  int ct = t / ksteps;
  int col  = (ct << 4) + (lane & 15);
  int base = (lane >= 16) ? 8 : 0;
  int k    = (kt << 5) + base + ((e < 8) ? e : e + 8);
  dst[idx] = f2bf(W[k * Fout + col]);
}

// ---------------------------------------------------------------------------
// fused (dual-)GEMM + bias [+ ReLU + LayerNorm] via v_wmma_f32_16x16x32_bf16
//   MODE_DUAL_LN: out = LN(relu(A0@B0 + A1@B1 + bias))   (GraphConv layer)
//   MODE_ADD_LN : out = LN(relu((A0+A1)@B0 + bias))      (fc layer)
//   MODE_DUAL   : out = A0@B0 + A1@B1 + bias             (mu / logstd heads)
// grid.x = N/16 (16 output rows per workgroup), one wave per 16-col tile.
// Whole 16 x FIN A-stripe staged in LDS once (fragment order, bf16), then a
// fully unrolled barrier-free K loop with two independent WMMA accumulators.
// ---------------------------------------------------------------------------
template <int FIN, int FOUT, int MODE>
__global__ void fused_gemm_wmma(const float* __restrict__ A0,
                                const float* __restrict__ A1,
                                const v16bf* __restrict__ B0,
                                const v16bf* __restrict__ B1,
                                const float* __restrict__ bias,
                                const float* __restrict__ gamma,
                                const float* __restrict__ beta,
                                float* __restrict__ out) {
  constexpr int  KSTEPS = FIN / 32;
  constexpr int  NW     = FOUT / 16;       // waves per block
  constexpr int  T      = NW * 32;         // threads per block
  constexpr bool DUAL   = (MODE != MODE_ADD_LN);
  constexpr bool LN     = (MODE != MODE_DUAL);

  __shared__ v16bf AsF[2][KSTEPS][32];     // bf16 A fragments (<= 16 KB)
  __shared__ float outTile[16][260];       // +4 pad kills bank conflicts

  const int lane    = threadIdx.x & 31;
  const int wave    = threadIdx.x >> 5;
  const int rowBase = blockIdx.x << 4;

  // ---- stage the whole 16 x FIN stripe, fp32 -> bf16, fragment order -------
  constexpr int QPR = FIN / 4;             // float4 quads per row
  constexpr int ITS = (16 * QPR) / T;      // always exact for our shapes
#pragma unroll
  for (int it = 0; it < ITS; ++it) {
    int q   = it * T + threadIdx.x;
    int row = q / QPR;
    int k   = (q - row * QPR) << 2;        // 4-aligned K within the row
    int kt  = k >> 5;
    int kk  = k & 31;
    int flane = row + (((kk >> 3) & 1) << 4);
    int felem = (kk & 7) + ((kk >> 4) << 3);
    size_t gi = (size_t)(rowBase + row) * FIN + k;
    float4 v0 = *reinterpret_cast<const float4*>(A0 + gi);
    float4 v1 = *reinterpret_cast<const float4*>(A1 + gi);
    __bf16* s0 = reinterpret_cast<__bf16*>(&AsF[0][kt][flane]) + felem;
    if (MODE == MODE_ADD_LN) {
      v0.x += v1.x; v0.y += v1.y; v0.z += v1.z; v0.w += v1.w;
      *reinterpret_cast<unsigned long long*>(s0) = pack4bf(v0);
    } else {
      *reinterpret_cast<unsigned long long*>(s0) = pack4bf(v0);
      __bf16* s1 = reinterpret_cast<__bf16*>(&AsF[1][kt][flane]) + felem;
      *reinterpret_cast<unsigned long long*>(s1) = pack4bf(v1);
    }
  }
  __syncthreads();                          // the only barrier before the LN epilogue

  // ---- barrier-free, fully unrolled K loop; two independent XDL chains ----
  v8f c0 = {0.f, 0.f, 0.f, 0.f, 0.f, 0.f, 0.f, 0.f};
  v8f c1 = {0.f, 0.f, 0.f, 0.f, 0.f, 0.f, 0.f, 0.f};
#pragma unroll
  for (int kt = 0; kt < KSTEPS; ++kt) {
    v16bf a0 = AsF[0][kt][lane];
    v16bf b0 = B0[(wave * KSTEPS + kt) * 32 + lane];
    c0 = __builtin_amdgcn_wmma_f32_16x16x32_bf16(false, a0, false, b0,
                                                 (short)0, c0, false, false);
    if (DUAL) {
      v16bf a1 = AsF[1][kt][lane];
      v16bf b1 = B1[(wave * KSTEPS + kt) * 32 + lane];
      c1 = __builtin_amdgcn_wmma_f32_16x16x32_bf16(false, a1, false, b1,
                                                   (short)0, c1, false, false);
    }
  }

  // C/D layout: elem r of lane l -> row r + (l>=16 ? 8 : 0), col = l & 15
  const int mhalf = (lane >= 16) ? 8 : 0;
  const int col   = (wave << 4) + (lane & 15);
  const float bv  = bias[col];

  float cf[8];
#pragma unroll
  for (int r = 0; r < 8; ++r)
    cf[r] = c0[r] + (DUAL ? c1[r] : 0.f) + bv;

  if (!LN) {   // mu / logstd heads: bias only, straight to out
#pragma unroll
    for (int r = 0; r < 8; ++r)
      out[(size_t)(rowBase + r + mhalf) * FOUT + col] = cf[r];
    return;
  }

  // ---- epilogue: ReLU into LDS, LayerNorm across FOUT == 256 --------------
#pragma unroll
  for (int r = 0; r < 8; ++r)
    outTile[r + mhalf][col] = cf[r] > 0.f ? cf[r] : 0.f;
  __syncthreads();

  const int row = wave;                     // 16 waves <-> 16 rows
  float s = 0.f, s2 = 0.f;
#pragma unroll
  for (int j = 0; j < 8; ++j) {
    float v = outTile[row][lane + (j << 5)];
    s += v; s2 += v * v;
  }
#pragma unroll
  for (int off = 16; off > 0; off >>= 1) {
    s  += __shfl_xor(s,  off, 32);
    s2 += __shfl_xor(s2, off, 32);
  }
  float mean = s * (1.f / 256.f);
  float var  = s2 * (1.f / 256.f) - mean * mean;
  float inv  = rsqrtf(var + 1e-5f);
  size_t orow = (size_t)(rowBase + row) * 256;
#pragma unroll
  for (int j = 0; j < 8; ++j) {
    int cc = lane + (j << 5);
    float v = outTile[row][cc];
    out[orow + cc] = (v - mean) * inv * gamma[cc] + beta[cc];
  }
}

// ---------------------------------------------------------------------------
// host orchestration
// ---------------------------------------------------------------------------
extern "C" void kernel_launch(void* const* d_in, const int* in_sizes, int n_in,
                              void* d_out, int out_size, void* d_ws, size_t ws_size,
                              hipStream_t stream) {
  const int N = NNODES;
  const float* x        = (const float*)d_in[0];
  const long long* ei   = (const long long*)d_in[1];
  const long long* esrc = ei;
  const long long* edst = ei + NEDGES;
  const float* Wrel1   = (const float*)d_in[2];
  const float* brel1   = (const float*)d_in[3];
  const float* Wroot1  = (const float*)d_in[4];
  const float* Wrel2   = (const float*)d_in[5];
  const float* brel2   = (const float*)d_in[6];
  const float* Wroot2  = (const float*)d_in[7];
  const float* Wfc     = (const float*)d_in[8];
  const float* bfc     = (const float*)d_in[9];
  const float* Wrel3   = (const float*)d_in[10];
  const float* brel3   = (const float*)d_in[11];
  const float* Wroot3  = (const float*)d_in[12];
  const float* Wrel_mu = (const float*)d_in[13];
  const float* brel_mu = (const float*)d_in[14];
  const float* Wroot_mu= (const float*)d_in[15];
  const float* Wrel_ls = (const float*)d_in[16];
  const float* brel_ls = (const float*)d_in[17];
  const float* Wroot_ls= (const float*)d_in[18];
  const float* g1 = (const float*)d_in[19]; const float* b1 = (const float*)d_in[20];
  const float* g2 = (const float*)d_in[21]; const float* b2 = (const float*)d_in[22];
  const float* g3 = (const float*)d_in[23]; const float* b3 = (const float*)d_in[24];
  const float* g4 = (const float*)d_in[25]; const float* b4 = (const float*)d_in[26];

  char* ws = (char*)d_ws;
  size_t off = 0;
  auto take = [&](size_t bytes) -> char* {
    char* p = ws + off;
    off += (bytes + 255) & ~(size_t)255;
    return p;
  };

  // bf16 fragment-packed weights (~0.9 MB total)
  __bf16* wbrel1  = (__bf16*)take((size_t)128 * 256 * 2);
  __bf16* wbroot1 = (__bf16*)take((size_t)128 * 256 * 2);
  __bf16* wbrel2  = (__bf16*)take((size_t)256 * 256 * 2);
  __bf16* wbroot2 = (__bf16*)take((size_t)256 * 256 * 2);
  __bf16* wbfc    = (__bf16*)take((size_t)256 * 256 * 2);
  __bf16* wbrel3  = (__bf16*)take((size_t)256 * 256 * 2);
  __bf16* wbroot3 = (__bf16*)take((size_t)256 * 256 * 2);
  __bf16* wbmu_r  = (__bf16*)take((size_t)256 * 64 * 2);
  __bf16* wbmu_o  = (__bf16*)take((size_t)256 * 64 * 2);
  __bf16* wbls_r  = (__bf16*)take((size_t)256 * 64 * 2);
  __bf16* wbls_o  = (__bf16*)take((size_t)256 * 64 * 2);

  // activation slots (4 x 51.2 MB)
  float* S0 = (float*)take((size_t)N * 256 * 4);   // aggregation scratch
  float* S1 = (float*)take((size_t)N * 256 * 4);   // x1, later x4
  float* S2 = (float*)take((size_t)N * 256 * 4);   // x2
  float* S3 = (float*)take((size_t)N * 256 * 4);   // x3

  auto pp = [&](const float* W, __bf16* dstp, int Fin, int Fout) {
    int total = Fin * Fout;
    prepack_w<<<dim3((total + 255) / 256), dim3(256), 0, stream>>>(W, dstp, Fin, Fout);
  };
  pp(Wrel1,  wbrel1, 128, 256);  pp(Wroot1, wbroot1, 128, 256);
  pp(Wrel2,  wbrel2, 256, 256);  pp(Wroot2, wbroot2, 256, 256);
  pp(Wfc,    wbfc,   256, 256);
  pp(Wrel3,  wbrel3, 256, 256);  pp(Wroot3, wbroot3, 256, 256);
  pp(Wrel_mu, wbmu_r, 256, 64);  pp(Wroot_mu, wbmu_o, 256, 64);
  pp(Wrel_ls, wbls_r, 256, 64);  pp(Wroot_ls, wbls_o, 256, 64);

  auto zero = [&](float* p, long long n) {
    long long n4 = n >> 2;
    zero_f4<<<dim3((unsigned)((n4 + 255) / 256)), dim3(256), 0, stream>>>((float4*)p, n4);
  };
  const dim3 gGrid(N / 16);   // 3125, exact

  // ---- layer 1: x1 = LN(relu(agg(x)@Wrel1 + brel1 + x@Wroot1)) ----
  zero(S0, (long long)N * 128);
  scatter_add4<128><<<dim3(NEDGES / 8), dim3(32, 8), 0, stream>>>(x, esrc, edst, S0);
  fused_gemm_wmma<128, 256, MODE_DUAL_LN><<<gGrid, dim3(512), 0, stream>>>(
      S0, x, (const v16bf*)wbrel1, (const v16bf*)wbroot1, brel1, g1, b1, S1);

  // ---- layer 2 ----
  zero(S0, (long long)N * 256);
  scatter_add4<256><<<dim3(NEDGES / 4), dim3(64, 4), 0, stream>>>(S1, esrc, edst, S0);
  fused_gemm_wmma<256, 256, MODE_DUAL_LN><<<gGrid, dim3(512), 0, stream>>>(
      S0, S1, (const v16bf*)wbrel2, (const v16bf*)wbroot2, brel2, g2, b2, S2);

  // ---- fc: x3 = LN(relu((x1+x2)@Wfc + bfc)) ----
  fused_gemm_wmma<256, 256, MODE_ADD_LN><<<gGrid, dim3(512), 0, stream>>>(
      S1, S2, (const v16bf*)wbfc, (const v16bf*)wbfc, bfc, g3, b3, S3);

  // ---- layer 3: x4 (reuse S1) ----
  zero(S0, (long long)N * 256);
  scatter_add4<256><<<dim3(NEDGES / 4), dim3(64, 4), 0, stream>>>(S3, esrc, edst, S0);
  fused_gemm_wmma<256, 256, MODE_DUAL_LN><<<gGrid, dim3(512), 0, stream>>>(
      S0, S3, (const v16bf*)wbrel3, (const v16bf*)wbroot3, brel3, g4, b4, S1);

  // ---- shared aggregation of x4 for both heads ----
  zero(S0, (long long)N * 256);
  scatter_add4<256><<<dim3(NEDGES / 4), dim3(64, 4), 0, stream>>>(S1, esrc, edst, S0);

  float* mu_out = (float*)d_out;
  float* ls_out = (float*)d_out + (size_t)N * 64;
  fused_gemm_wmma<256, 64, MODE_DUAL><<<gGrid, dim3(128), 0, stream>>>(
      S0, S1, (const v16bf*)wbmu_r, (const v16bf*)wbmu_o, brel_mu, nullptr, nullptr, mu_out);
  fused_gemm_wmma<256, 64, MODE_DUAL><<<gGrid, dim3(128), 0, stream>>>(
      S0, S1, (const v16bf*)wbls_r, (const v16bf*)wbls_o, brel_ls, nullptr, nullptr, ls_out);

  (void)in_sizes; (void)n_in; (void)out_size; (void)ws_size;
}
